// TNO_3298534883789
// MI455X (gfx1250) — matrix operand order repaired and checked by reading, not compiled, not run on hardware
//
#include <hip/hip_runtime.h>

// ---------------------------------------------------------------------------
// TNO layer for MI455X (gfx1250, wave32, WMMA bf16 16x16x32)
// Pipeline: rmsnorm -> u,v GEMM(+silu) -> causal Toeplitz (masked WMMA GEMM,
// T tiles synthesized from LDS-cached t) -> g=u*y -> out GEMM + bias + resid.
// GEMMs: 256-thr block = 8 waves, wave = 32x64 strip (8 accums), B tile staged
// in LDS via gfx1250 async global->LDS copies, double buffered.
// ---------------------------------------------------------------------------

#define SEQ      2048
#define BATCH    4
#define EMBED    1024
#define D1       2048
#define HEADS    8
#define HEAD_DIM 256
#define ROWS     (SEQ * BATCH)   // 8192 flattened (i,b) rows

#define BROW     40              // LDS B-tile row stride in ushorts (80B, 16B-mult)

typedef __attribute__((ext_vector_type(16))) __bf16 v16bf;
typedef __attribute__((ext_vector_type(8)))  float  v8f;
typedef int v4i32 __attribute__((vector_size(16)));   // matches builtin proto

union FragB16 { v16bf v; uint4 q[2]; unsigned short s[16]; };

#if defined(__has_builtin)
#if __has_builtin(__builtin_amdgcn_global_load_async_to_lds_b128) && \
    __has_builtin(__builtin_amdgcn_s_wait_asynccnt)
#define HAVE_ASYNC_LDS 1
#endif
#endif
#ifndef HAVE_ASYNC_LDS
#define HAVE_ASYNC_LDS 0
#endif

__device__ __forceinline__ void async_copy16(const unsigned short* g,
                                             unsigned short* l) {
#if HAVE_ASYNC_LDS
    __builtin_amdgcn_global_load_async_to_lds_b128(
        (__attribute__((address_space(1))) v4i32*)g,
        (__attribute__((address_space(3))) v4i32*)l, 0, 0);
#else
    *(uint4*)l = *(const uint4*)g;
#endif
}
__device__ __forceinline__ void async_wait0() {
#if HAVE_ASYNC_LDS
    __builtin_amdgcn_s_wait_asynccnt(0);
#endif
}

__device__ __forceinline__ unsigned short f2bf(float f) {
    union { float f; unsigned u; } x; x.f = f;
    unsigned r = x.u + 0x7FFFu + ((x.u >> 16) & 1u);   // round-to-nearest-even
    return (unsigned short)(r >> 16);
}
__device__ __forceinline__ float bf2f(unsigned short h) {
    union { unsigned u; float f; } x; x.u = ((unsigned)h) << 16;
    return x.f;
}
__device__ __forceinline__ float silu(float x) {
    return x / (1.0f + __expf(-x));
}
__device__ __forceinline__ v8f wmma_bf16(const FragB16& a, const FragB16& b,
                                         v8f c) {
    return __builtin_amdgcn_wmma_f32_16x16x32_bf16(
        false, a.v, false, b.v, (short)0, c, false, false);
}

// ---------------------------------------------------------------------------
// f32 -> bf16 bulk conversion (n multiple of 4)
// ---------------------------------------------------------------------------
__global__ void k_f32_to_bf16(const float* __restrict__ src,
                              unsigned short* __restrict__ dst, int n) {
    int i = (blockIdx.x * blockDim.x + threadIdx.x) * 4;
    if (i + 3 < n) {
        float4 v = *(const float4*)(src + i);
        dst[i + 0] = f2bf(v.x);
        dst[i + 1] = f2bf(v.y);
        dst[i + 2] = f2bf(v.z);
        dst[i + 3] = f2bf(v.w);
    }
}

// ---------------------------------------------------------------------------
// Build t[h][0..SEQ-1] = [t_zero[h], t_pos[h][0..SEQ-2]] in bf16
// ---------------------------------------------------------------------------
__global__ void k_build_t(const float* __restrict__ tz,
                          const float* __restrict__ tp,
                          unsigned short* __restrict__ tbf) {
    int idx = blockIdx.x * blockDim.x + threadIdx.x;   // 0 .. HEADS*SEQ-1
    int h = idx >> 11, i = idx & (SEQ - 1);
    float val = (i == 0) ? tz[h] : tp[h * (SEQ - 1) + i - 1];
    tbf[idx] = f2bf(val);
}

// ---------------------------------------------------------------------------
// SimpleRMSNorm: one block (256 thr) per row of 1024 floats -> bf16
// ---------------------------------------------------------------------------
__global__ void k_rmsnorm(const float* __restrict__ x,
                          unsigned short* __restrict__ xn) {
    __shared__ float red[8];
    const int row = blockIdx.x;
    const float* xr = x + (size_t)row * EMBED;
    const int tid = threadIdx.x;
    float4 v = *(const float4*)(xr + tid * 4);
    float ss = v.x * v.x + v.y * v.y + v.z * v.z + v.w * v.w;
    #pragma unroll
    for (int o = 16; o > 0; o >>= 1) ss += __shfl_xor(ss, o, 32);
    if ((tid & 31) == 0) red[tid >> 5] = ss;
    __syncthreads();
    if (tid == 0) {
        float t = 0.f;
        #pragma unroll
        for (int w = 0; w < 8; ++w) t += red[w];
        red[0] = t;
    }
    __syncthreads();
    float total = red[0];
    float rms = sqrtf(total) * (1.0f / 32.0f);     // ||x|| * 1024^{-1/2}
    float inv = 1.0f / (rms + 1e-8f);
    unsigned short* o = xn + (size_t)row * EMBED + tid * 4;
    o[0] = f2bf(v.x * inv); o[1] = f2bf(v.y * inv);
    o[2] = f2bf(v.z * inv); o[3] = f2bf(v.w * inv);
}

// ---------------------------------------------------------------------------
// u/v projection GEMM: C(8192x2048) = silu(XN(8192x1024) * W(2048x1024)^T + b)
// blockIdx.z: 0 -> u (row-major), 1 -> v (stored transposed (b,h,d,seq)).
// Block: 256 rows x 64 cols; wave: 32x64. B tile async-staged in LDS (x2 buf).
// ---------------------------------------------------------------------------
__global__ __launch_bounds__(256)
void k_uv_gemm(const unsigned short* __restrict__ xn,
               const unsigned short* __restrict__ wu,
               const unsigned short* __restrict__ wvv,
               const float* __restrict__ ub,
               const float* __restrict__ vb,
               unsigned short* __restrict__ uo,
               unsigned short* __restrict__ vt) {
    __shared__ unsigned short ldsB[2][64 * BROW];
    const int tid = threadIdx.x;
    const int lane = tid & 31, wid = tid >> 5;
    const int sel = lane >> 4, l15 = lane & 15;
    const int m0 = blockIdx.y * 256 + wid * 32;
    const int n_base = blockIdx.x * 64;
    const bool isV = (blockIdx.z != 0);
    const unsigned short* W = isV ? wvv : wu;
    const float* Bb = isV ? vb : ub;
    const int Kd = EMBED, KT = EMBED / 32;

    const int nl = tid >> 2, ch = tid & 3;     // 64 rows x 4 chunks of 16B
    auto stageB = [&](int kt, int buf) {
        const unsigned short* src = W + (size_t)(n_base + nl) * Kd + kt * 32 + ch * 8;
        async_copy16(src, &ldsB[buf][nl * BROW + ch * 8]);
    };
    auto loadA = [&](FragB16* A, int k0) {
        #pragma unroll
        for (int mi = 0; mi < 2; ++mi) {
            const unsigned short* ar =
                xn + (size_t)(m0 + mi * 16 + l15) * Kd + k0 + sel * 8;
            A[mi].q[0] = *(const uint4*)(ar);
            A[mi].q[1] = *(const uint4*)(ar + 16);
        }
    };

    v8f acc[2][4];
    #pragma unroll
    for (int mi = 0; mi < 2; ++mi)
        #pragma unroll
        for (int t = 0; t < 4; ++t) acc[mi][t] = (v8f){0,0,0,0,0,0,0,0};

    FragB16 a[2];
    stageB(0, 0);
    loadA(a, 0);
    for (int kt = 0; kt < KT; ++kt) {
        const int buf = kt & 1;
        async_wait0();
        __syncthreads();                       // tile kt resident, prev reads done
        const bool more = (kt + 1 < KT);
        FragB16 a2[2];
        if (more) { stageB(kt + 1, buf ^ 1); loadA(a2, (kt + 1) * 32); }
        #pragma unroll
        for (int t = 0; t < 4; ++t) {
            const unsigned short* bp = &ldsB[buf][(t * 16 + l15) * BROW + sel * 8];
            FragB16 b;
            b.q[0] = *(const uint4*)(bp);
            b.q[1] = *(const uint4*)(bp + 16);
            acc[0][t] = wmma_bf16(a[0], b, acc[0][t]);
            acc[1][t] = wmma_bf16(a[1], b, acc[1][t]);
        }
        if (more) { a[0] = a2[0]; a[1] = a2[1]; }
    }

    #pragma unroll
    for (int t = 0; t < 4; ++t) {
        const int n = n_base + t * 16 + l15;
        const float bias = Bb[n];
        #pragma unroll
        for (int mi = 0; mi < 2; ++mi) {
            #pragma unroll
            for (int r = 0; r < 8; ++r) {
                const int m = m0 + mi * 16 + sel * 8 + r;
                const unsigned short hv = f2bf(silu(acc[mi][t][r] + bias));
                if (!isV) {
                    uo[(size_t)m * D1 + n] = hv;
                } else {
                    const int i = m >> 2, bb = m & 3;
                    const int h = n >> 8, dd = n & 255;
                    vt[((size_t)((bb * HEADS + h) * HEAD_DIM) + dd) * SEQ + i] = hv;
                }
            }
        }
    }
}

// ---------------------------------------------------------------------------
// Causal Toeplitz as masked WMMA GEMM + fused gate g = u * y.
// per (b,h): Y(2048x256) = T(lower-tri Toeplitz) * V(2048x256).
// T fragments synthesized per-lane from LDS-cached t vector (causal via d>=0).
// Block-uniform K bound (barriers inside loop); excess tiles masked to zero.
// grid = (HEAD_DIM/64, SEQ/256, 32), block = 256.
// ---------------------------------------------------------------------------
__global__ __launch_bounds__(256)
void k_toeplitz(const unsigned short* __restrict__ vt,
                const unsigned short* __restrict__ tbf,
                const unsigned short* __restrict__ u,
                unsigned short* __restrict__ g) {
    __shared__ unsigned short t_s[SEQ];
    __shared__ unsigned short ldsB[2][64 * BROW];
    const int z = blockIdx.z;               // 0..31 = (b,h)
    const int bb = z >> 3, h = z & 7;
    for (int i = threadIdx.x; i < SEQ; i += blockDim.x)
        t_s[i] = tbf[h * SEQ + i];

    const int tid = threadIdx.x;
    const int lane = tid & 31, wid = tid >> 5;
    const int sel = lane >> 4, l15 = lane & 15;
    const int m0 = (blockIdx.y * 8 + wid) * 32;   // seq row strip (32 rows)
    const int n_base = blockIdx.x * 64;           // head-dim cols
    const unsigned short* vbase = vt + (size_t)z * HEAD_DIM * SEQ;
    const int KT = blockIdx.y * 8 + 8;            // block-uniform causal bound

    const int nl = tid >> 2, ch = tid & 3;
    auto stageB = [&](int kt, int buf) {
        const unsigned short* src =
            vbase + (size_t)(n_base + nl) * SEQ + kt * 32 + ch * 8;
        async_copy16(src, &ldsB[buf][nl * BROW + ch * 8]);
    };

    v8f acc[2][4];
    #pragma unroll
    for (int mi = 0; mi < 2; ++mi)
        #pragma unroll
        for (int t = 0; t < 4; ++t) acc[mi][t] = (v8f){0,0,0,0,0,0,0,0};

    stageB(0, 0);
    for (int kt = 0; kt < KT; ++kt) {
        const int buf = kt & 1;
        async_wait0();
        __syncthreads();                    // also covers initial t_s fill
        if (kt + 1 < KT) stageB(kt + 1, buf ^ 1);
        const int k0 = kt * 32;
        FragB16 a[2];
        #pragma unroll
        for (int mi = 0; mi < 2; ++mi) {
            const int mrow = m0 + mi * 16 + l15;
            #pragma unroll
            for (int j = 0; j < 8; ++j) {
                const int d0 = mrow - (k0 + sel * 8 + j);       // T[m][k]=t[m-k]
                const int d1 = mrow - (k0 + 16 + sel * 8 + j);
                a[mi].s[j]     = (d0 >= 0) ? t_s[d0] : (unsigned short)0;
                a[mi].s[j + 8] = (d1 >= 0) ? t_s[d1] : (unsigned short)0;
            }
        }
        #pragma unroll
        for (int t = 0; t < 4; ++t) {
            const unsigned short* bp = &ldsB[buf][(t * 16 + l15) * BROW + sel * 8];
            FragB16 b;
            b.q[0] = *(const uint4*)(bp);
            b.q[1] = *(const uint4*)(bp + 16);
            acc[0][t] = wmma_bf16(a[0], b, acc[0][t]);
            acc[1][t] = wmma_bf16(a[1], b, acc[1][t]);
        }
    }

    #pragma unroll
    for (int t = 0; t < 4; ++t) {
        const int dd = n_base + t * 16 + l15;
        const int col = h * HEAD_DIM + dd;
        #pragma unroll
        for (int mi = 0; mi < 2; ++mi) {
            #pragma unroll
            for (int r = 0; r < 8; ++r) {
                const int i = m0 + mi * 16 + sel * 8 + r;
                const size_t idx = (size_t)(i * BATCH + bb) * D1 + col;
                g[idx] = f2bf(bf2f(u[idx]) * acc[mi][t][r]);   // g = u * y
            }
        }
    }
}

// ---------------------------------------------------------------------------
// Output GEMM: out(8192x1024) = G(8192x2048) * Wo(1024x2048)^T + ob + x
// ---------------------------------------------------------------------------
__global__ __launch_bounds__(256)
void k_out_gemm(const unsigned short* __restrict__ g,
                const unsigned short* __restrict__ wo,
                const float* __restrict__ ob,
                const float* __restrict__ x,
                float* __restrict__ out) {
    __shared__ unsigned short ldsB[2][64 * BROW];
    const int tid = threadIdx.x;
    const int lane = tid & 31, wid = tid >> 5;
    const int sel = lane >> 4, l15 = lane & 15;
    const int m0 = blockIdx.y * 256 + wid * 32;
    const int n_base = blockIdx.x * 64;
    const int Kd = D1, KT = D1 / 32;

    const int nl = tid >> 2, ch = tid & 3;
    auto stageB = [&](int kt, int buf) {
        const unsigned short* src = wo + (size_t)(n_base + nl) * Kd + kt * 32 + ch * 8;
        async_copy16(src, &ldsB[buf][nl * BROW + ch * 8]);
    };
    auto loadA = [&](FragB16* A, int k0) {
        #pragma unroll
        for (int mi = 0; mi < 2; ++mi) {
            const unsigned short* ar =
                g + (size_t)(m0 + mi * 16 + l15) * Kd + k0 + sel * 8;
            A[mi].q[0] = *(const uint4*)(ar);
            A[mi].q[1] = *(const uint4*)(ar + 16);
        }
    };

    v8f acc[2][4];
    #pragma unroll
    for (int mi = 0; mi < 2; ++mi)
        #pragma unroll
        for (int t = 0; t < 4; ++t) acc[mi][t] = (v8f){0,0,0,0,0,0,0,0};

    FragB16 a[2];
    stageB(0, 0);
    loadA(a, 0);
    for (int kt = 0; kt < KT; ++kt) {
        const int buf = kt & 1;
        async_wait0();
        __syncthreads();
        const bool more = (kt + 1 < KT);
        FragB16 a2[2];
        if (more) { stageB(kt + 1, buf ^ 1); loadA(a2, (kt + 1) * 32); }
        #pragma unroll
        for (int t = 0; t < 4; ++t) {
            const unsigned short* bp = &ldsB[buf][(t * 16 + l15) * BROW + sel * 8];
            FragB16 b;
            b.q[0] = *(const uint4*)(bp);
            b.q[1] = *(const uint4*)(bp + 16);
            acc[0][t] = wmma_bf16(a[0], b, acc[0][t]);
            acc[1][t] = wmma_bf16(a[1], b, acc[1][t]);
        }
        if (more) { a[0] = a2[0]; a[1] = a2[1]; }
    }

    #pragma unroll
    for (int t = 0; t < 4; ++t) {
        const int n = n_base + t * 16 + l15;
        const float bias = ob[n];
        #pragma unroll
        for (int mi = 0; mi < 2; ++mi) {
            #pragma unroll
            for (int r = 0; r < 8; ++r) {
                const int m = m0 + mi * 16 + sel * 8 + r;
                const size_t idx = (size_t)m * EMBED + n;
                out[idx] = acc[mi][t][r] + bias + x[idx];
            }
        }
    }
}

// ---------------------------------------------------------------------------
extern "C" void kernel_launch(void* const* d_in, const int* in_sizes, int n_in,
                              void* d_out, int out_size, void* d_ws, size_t ws_size,
                              hipStream_t stream) {
    (void)in_sizes; (void)n_in; (void)out_size; (void)ws_size;

    const float* x      = (const float*)d_in[0];
    const float* u_w    = (const float*)d_in[1];
    const float* u_b    = (const float*)d_in[2];
    const float* v_w    = (const float*)d_in[3];
    const float* v_b    = (const float*)d_in[4];
    const float* o_w    = (const float*)d_in[5];
    const float* o_b    = (const float*)d_in[6];
    const float* t_zero = (const float*)d_in[7];
    const float* t_pos  = (const float*)d_in[8];
    float* out = (float*)d_out;

    char* ws = (char*)d_ws;
    size_t off = 0;
    auto carve = [&](size_t bytes) {
        void* p = ws + off;
        off += (bytes + 255) & ~(size_t)255;
        return p;
    };
    unsigned short* WU  = (unsigned short*)carve((size_t)D1 * EMBED * 2);
    unsigned short* WV  = (unsigned short*)carve((size_t)D1 * EMBED * 2);
    unsigned short* WO  = (unsigned short*)carve((size_t)EMBED * D1 * 2);
    unsigned short* TBF = (unsigned short*)carve((size_t)HEADS * SEQ * 2);
    unsigned short* XN  = (unsigned short*)carve((size_t)ROWS * EMBED * 2);
    unsigned short* U   = (unsigned short*)carve((size_t)ROWS * D1 * 2);
    unsigned short* VT  = (unsigned short*)carve((size_t)ROWS * D1 * 2);
    unsigned short* G   = (unsigned short*)carve((size_t)ROWS * D1 * 2);

    const int NW = D1 * EMBED;                       // 2,097,152
    k_f32_to_bf16<<<NW / 1024, 256, 0, stream>>>(u_w, WU, NW);
    k_f32_to_bf16<<<NW / 1024, 256, 0, stream>>>(v_w, WV, NW);
    k_f32_to_bf16<<<NW / 1024, 256, 0, stream>>>(o_w, WO, NW);
    k_build_t<<<(HEADS * SEQ) / 256, 256, 0, stream>>>(t_zero, t_pos, TBF);
    k_rmsnorm<<<ROWS, 256, 0, stream>>>(x, XN);

    k_uv_gemm<<<dim3(D1 / 64, ROWS / 256, 2), 256, 0, stream>>>(
        XN, WU, WV, u_b, v_b, U, VT);

    k_toeplitz<<<dim3(HEAD_DIM / 64, SEQ / 256, BATCH * HEADS), 256, 0, stream>>>(
        VT, TBF, U, G);

    k_out_gemm<<<dim3(EMBED / 64, ROWS / 256), 256, 0, stream>>>(
        G, WO, o_b, x, out);
}
